// ReprogrammingLayer_37598143709495
// MI455X (gfx1250) — matrix operand order repaired and checked by reading, not compiled, not run on hardware
//
#include <hip/hip_runtime.h>

typedef __attribute__((ext_vector_type(16))) __bf16 v16bf;
typedef __attribute__((ext_vector_type(8)))  __bf16 v8bf;
typedef __attribute__((ext_vector_type(4)))  __bf16 v4bf;
typedef __attribute__((ext_vector_type(8)))  float  v8f;

// ---------------- helpers ----------------

__device__ __forceinline__ void load4(const float* p, float (&d)[4]) {
  float4 t = *(const float4*)p;
  d[0] = t.x; d[1] = t.y; d[2] = t.z; d[3] = t.w;
}
__device__ __forceinline__ void load4(const __bf16* p, float (&d)[4]) {
  v4bf t = *(const v4bf*)p;
#pragma unroll
  for (int i = 0; i < 4; ++i) d[i] = (float)t[i];
}

__device__ __forceinline__ v16bf make_frag(const __bf16* lo_p, const __bf16* hi_p) {
  v8bf lo = *(const v8bf*)lo_p;
  v8bf hi = *(const v8bf*)hi_p;
  v16bf r;
#pragma unroll
  for (int i = 0; i < 8; ++i) { r[i] = lo[i]; r[i + 8] = hi[i]; }
  return r;
}

// A-operand fragment (16x32 bf16) from a row-major [m][k] buffer.
// lane row = lane&15; halves at K = k0+(lane>>4)*8 and +16.
__device__ __forceinline__ v16bf frag_a(const __bf16* row, int k0, int lane) {
  int kb = k0 + ((lane >> 4) << 3);
  return make_frag(row + kb, row + kb + 16);
}

// B-operand fragment (32x16 bf16) from an [n][k] (transposed) buffer.
// lane col = lane&15; halves at K = k0+(lane>>4)*8 and +16.
__device__ __forceinline__ v16bf frag_b(const __bf16* base, int ldn, int n0, int k0, int lane) {
  const __bf16* p = base + (size_t)(n0 + (lane & 15)) * ldn + k0 + ((lane >> 4) << 3);
  return make_frag(p, p + 16);
}

__device__ __forceinline__ v8f wmma_bf16(v16bf a, v16bf b, v8f c) {
  return __builtin_amdgcn_wmma_f32_16x16x32_bf16(false, a, false, b, (short)0, c,
                                                 false, false);
}

// Async global -> LDS copy, 16B per lane (GLOBAL_LOAD_ASYNC_TO_LDS_B128, ASYNCcnt).
__device__ __forceinline__ void async_copy_b128(const void* gaddr, void* lds_ptr) {
  unsigned lds_off = (unsigned)(__SIZE_TYPE__)lds_ptr;  // addrspace(3) offset
  asm volatile("global_load_async_to_lds_b128 %0, %1, off"
               :
               : "v"(lds_off), "v"(gaddr)
               : "memory");
}
__device__ __forceinline__ void wait_asynccnt0() {
#if __has_builtin(__builtin_amdgcn_s_wait_asynccnt)
  __builtin_amdgcn_s_wait_asynccnt(0);
#else
  asm volatile("s_wait_asynccnt 0x0" ::: "memory");
#endif
}

// ---------------- tiled WMMA GEMM with bias: C = (A@B + bias) * out_scale ----------------
// A: [Mpad,K] (AT = float or __bf16, rows >= M_A read as 0), B: [K,N] f32 row-major.
// Block = 128x128 tile, 8 waves in 4(M) x 2(N), each wave 32x64 via 2x4 WMMA tiles.
// Register-staged double buffering: global loads for tile t+1 overlap WMMAs of tile t.

template <typename AT, bool OUT_BF16>
__global__ __launch_bounds__(256) void gemm_bias_kernel(
    const AT* __restrict__ A, const float* __restrict__ B,
    const float* __restrict__ bias, void* __restrict__ Cout,
    int M_A, int K, int N, float out_scale) {
  constexpr int BM = 128, BN = 128, BK = 32;
  constexpr int LDA = BK + 8;  // 80B rows, 16B aligned
  constexpr int LDB = BK + 8;
  __shared__ __bf16 As[BM * LDA];   // [m][k]
  __shared__ __bf16 Bs[BN * LDB];   // [n][k]  (transposed -> B-operand layout)

  const int tid = threadIdx.x, lane = tid & 31, wid = tid >> 5;
  const int wm = wid >> 1, wn = wid & 1;
  const int m0 = blockIdx.y * BM, n0 = blockIdx.x * BN;

  v8f acc[2][4];
#pragma unroll
  for (int mt = 0; mt < 2; ++mt)
#pragma unroll
    for (int nt = 0; nt < 4; ++nt)
#pragma unroll
      for (int r = 0; r < 8; ++r) acc[mt][nt][r] = 0.0f;

  float areg[4][4], breg[4][4];

  auto load_tile = [&](int k0) {
    // A tile 128x32: 16B chunks, 4 per thread
#pragma unroll
    for (int i = 0; i < 4; ++i) {
      int idx4 = tid + i * 256;
      int m = idx4 >> 3, k4 = (idx4 & 7) << 2;
      int gm = m0 + m;
      if (gm < M_A) {
        load4(&A[(size_t)gm * K + k0 + k4], areg[i]);
      } else {
#pragma unroll
        for (int j = 0; j < 4; ++j) areg[i][j] = 0.0f;
      }
    }
    // B tile 32x128: 16B chunks, 4 per thread
#pragma unroll
    for (int i = 0; i < 4; ++i) {
      int idx4 = tid + i * 256;
      int k = idx4 >> 5, n4 = (idx4 & 31) << 2;
      load4(&B[(size_t)(k0 + k) * N + n0 + n4], breg[i]);
    }
  };

  auto store_tile = [&]() {
#pragma unroll
    for (int i = 0; i < 4; ++i) {
      int idx4 = tid + i * 256;
      int m = idx4 >> 3, k4 = (idx4 & 7) << 2;
      v4bf a;
#pragma unroll
      for (int j = 0; j < 4; ++j) a[j] = (__bf16)areg[i][j];
      *(v4bf*)&As[m * LDA + k4] = a;
    }
#pragma unroll
    for (int i = 0; i < 4; ++i) {
      int idx4 = tid + i * 256;
      int k = idx4 >> 5, n4 = (idx4 & 31) << 2;
#pragma unroll
      for (int j = 0; j < 4; ++j) Bs[(n4 + j) * LDB + k] = (__bf16)breg[i][j];
    }
  };

  load_tile(0);
  for (int k0 = 0; k0 < K; k0 += BK) {
    __syncthreads();
    store_tile();
    __syncthreads();
    if (k0 + BK < K) load_tile(k0 + BK);  // overlaps with WMMAs below

    v16bf af[2], bf[4];
    const int arow = lane & 15;
    af[0] = frag_a(&As[(wm * 32 + arow) * LDA], 0, lane);
    af[1] = frag_a(&As[(wm * 32 + 16 + arow) * LDA], 0, lane);
#pragma unroll
    for (int nt = 0; nt < 4; ++nt)
      bf[nt] = frag_b(Bs, LDB, wn * 64 + nt * 16, 0, lane);
#pragma unroll
    for (int mt = 0; mt < 2; ++mt)
#pragma unroll
      for (int nt = 0; nt < 4; ++nt)
        acc[mt][nt] = wmma_bf16(af[mt], bf[nt], acc[mt][nt]);
  }

  // epilogue: + bias, * out_scale, store (C/D layout: row=(lane>>4)*8+r, col=lane&15)
#pragma unroll
  for (int nt = 0; nt < 4; ++nt) {
    const int gn = n0 + wn * 64 + nt * 16 + (lane & 15);
    const float bv = bias[gn];
#pragma unroll
    for (int mt = 0; mt < 2; ++mt) {
      const int rbase = m0 + wm * 32 + mt * 16 + ((lane >> 4) << 3);
#pragma unroll
      for (int r = 0; r < 8; ++r) {
        float v = (acc[mt][nt][r] + bv) * out_scale;
        if (OUT_BF16)
          ((__bf16*)Cout)[(size_t)(rbase + r) * N + gn] = (__bf16)v;
        else
          ((float*)Cout)[(size_t)(rbase + r) * N + gn] = v;
      }
    }
  }
}

// ---------------- flash attention over one (b,h,128-row L tile) ----------------
// Q: [8192,1024] bf16 (already scaled by 1/sqrt(E)); K,V: [1024,1024] bf16 (S padded,
// pad rows hold the bias); O: [8192,1024] bf16. 8 waves, each owns 16 query rows.
// K tile is DMA'd into LDS with GLOBAL_LOAD_ASYNC_TO_LDS_B128 (ASYNCcnt path).

__global__ __launch_bounds__(256) void attention_kernel(
    const __bf16* __restrict__ Q, const __bf16* __restrict__ Kb,
    const __bf16* __restrict__ Vb, __bf16* __restrict__ O) {
  constexpr int ST = 64, SVALID = 1000, NT = 16;
  constexpr int LDK = 128 + 8;  // K tile [s][e]
  constexpr int LDV = ST + 8;   // V tile [e][s]
  constexpr int LDP = ST + 8;   // P tile per wave [row][s]
  __shared__ __bf16 Ks[ST * LDK];
  __shared__ __bf16 Vs[128 * LDV];
  __shared__ __bf16 Ps[8 * 16 * LDP];

  const int tid = threadIdx.x, lane = tid & 31, wid = tid >> 5;
  const int blk = blockIdx.x;
  const int lt = blk & 7, h = (blk >> 3) & 7, b = blk >> 6;
  const size_t qrow0 = (size_t)b * 1024 + lt * 128;

  // Q fragments held in registers for the whole S loop (4 k-steps of 32 over E=128)
  const int arow = lane & 15;
  v16bf qf[4];
  {
    const __bf16* qp = Q + (qrow0 + wid * 16 + arow) * 1024 + h * 128;
#pragma unroll
    for (int es = 0; es < 4; ++es) qf[es] = frag_a(qp, es * 32, lane);
  }

  v8f Oa[8];
  float m_r[8], l_r[8];
#pragma unroll
  for (int et = 0; et < 8; ++et)
#pragma unroll
    for (int r = 0; r < 8; ++r) Oa[et][r] = 0.0f;
#pragma unroll
  for (int r = 0; r < 8; ++r) { m_r[r] = -1e30f; l_r[r] = 0.0f; }

  for (int st = 0; st < NT; ++st) {
    const int s0 = st * ST;
    __syncthreads();
    // K tile [64][128]: async DMA straight into LDS (16B per lane, 4 chunks/thread)
#pragma unroll
    for (int i = 0; i < 4; ++i) {
      int idx = tid + i * 256;          // 8 bf16 per op
      int s = idx >> 4, e8 = (idx & 15) << 3;
      async_copy_b128(Kb + (size_t)(s0 + s) * 1024 + h * 128 + e8,
                      &Ks[s * LDK + e8]);
    }
    // V tile: through VGPRs, written transposed [e][s]
#pragma unroll
    for (int i = 0; i < 4; ++i) {
      int idx = tid + i * 256;
      int s = idx >> 4, e8 = (idx & 15) << 3;
      v8bf vv = *(const v8bf*)(Vb + (size_t)(s0 + s) * 1024 + h * 128 + e8);
#pragma unroll
      for (int j = 0; j < 8; ++j) Vs[(e8 + j) * LDV + s] = vv[j];
    }
    // prefetch next S tile of K/V into cache (global_prefetch_b8)
    if (st + 1 < NT) {
      const size_t prow = (size_t)(s0 + ST + (tid >> 2)) * 1024 + h * 128;
      const int pofs = (tid & 3) * 32;
      __builtin_prefetch(Kb + prow + pofs, 0, 0);
      __builtin_prefetch(Vb + prow + pofs, 0, 0);
    }
    wait_asynccnt0();
    __syncthreads();

    // scores: S[16 x 64] = Q[16x128] @ K^T  (4 n-tiles, 4 k-steps over E)
    v8f sc[4];
#pragma unroll
    for (int nt = 0; nt < 4; ++nt)
#pragma unroll
      for (int r = 0; r < 8; ++r) sc[nt][r] = 0.0f;
#pragma unroll
    for (int es = 0; es < 4; ++es) {
      v16bf bfr[4];
#pragma unroll
      for (int nt = 0; nt < 4; ++nt)
        bfr[nt] = frag_b(Ks, LDK, nt * 16, es * 32, lane);
#pragma unroll
      for (int nt = 0; nt < 4; ++nt)
        sc[nt] = wmma_bf16(qf[es], bfr[nt], sc[nt]);
    }

    // mask invalid cols, tile row-max
    float tm[8];
#pragma unroll
    for (int r = 0; r < 8; ++r) tm[r] = -1e30f;
#pragma unroll
    for (int nt = 0; nt < 4; ++nt) {
      const bool valid = (s0 + nt * 16 + (lane & 15)) < SVALID;
#pragma unroll
      for (int r = 0; r < 8; ++r) {
        float v = valid ? sc[nt][r] : -1e30f;
        sc[nt][r] = v;
        tm[r] = fmaxf(tm[r], v);
      }
    }
#pragma unroll
    for (int mask = 1; mask < 16; mask <<= 1)
#pragma unroll
      for (int r = 0; r < 8; ++r) tm[r] = fmaxf(tm[r], __shfl_xor(tm[r], mask, 32));

    float alpha[8], ts[8];
#pragma unroll
    for (int r = 0; r < 8; ++r) {
      float mn = fmaxf(m_r[r], tm[r]);
      alpha[r] = __expf(m_r[r] - mn);
      m_r[r] = mn;
      ts[r] = 0.0f;
    }
#pragma unroll
    for (int nt = 0; nt < 4; ++nt)
#pragma unroll
      for (int r = 0; r < 8; ++r) {
        float p = __expf(sc[nt][r] - m_r[r]);
        sc[nt][r] = p;
        ts[r] += p;
      }
#pragma unroll
    for (int mask = 1; mask < 16; mask <<= 1)
#pragma unroll
      for (int r = 0; r < 8; ++r) ts[r] += __shfl_xor(ts[r], mask, 32);
#pragma unroll
    for (int r = 0; r < 8; ++r) l_r[r] = l_r[r] * alpha[r] + ts[r];
#pragma unroll
    for (int et = 0; et < 8; ++et)
#pragma unroll
      for (int r = 0; r < 8; ++r) Oa[et][r] *= alpha[r];

    // P -> per-wave LDS region (A-operand row-major layout), same-wave DS is in-order
    __bf16* pw = &Ps[wid * 16 * LDP];
#pragma unroll
    for (int nt = 0; nt < 4; ++nt)
#pragma unroll
      for (int r = 0; r < 8; ++r)
        pw[(((lane >> 4) << 3) + r) * LDP + nt * 16 + (lane & 15)] = (__bf16)sc[nt][r];

    // O += P[16x64] @ V[64x128]  (2 k-steps over s, 8 e-tiles)
#pragma unroll
    for (int ks = 0; ks < 2; ++ks) {
      v16bf pf = frag_a(&pw[arow * LDP], ks * 32, lane);
      v16bf vfr[8];
#pragma unroll
      for (int et = 0; et < 8; ++et)
        vfr[et] = frag_b(Vs, LDV, et * 16, ks * 32, lane);
#pragma unroll
      for (int et = 0; et < 8; ++et)
        Oa[et] = wmma_bf16(pf, vfr[et], Oa[et]);
    }
  }

  // normalize and store
  const size_t orow0 = qrow0 + wid * 16 + ((lane >> 4) << 3);
#pragma unroll
  for (int r = 0; r < 8; ++r) l_r[r] = 1.0f / l_r[r];
#pragma unroll
  for (int et = 0; et < 8; ++et) {
    const int oc = h * 128 + et * 16 + (lane & 15);
#pragma unroll
    for (int r = 0; r < 8; ++r)
      O[(orow0 + r) * 1024 + oc] = (__bf16)(Oa[et][r] * l_r[r]);
  }
}

// ---------------- launcher ----------------

extern "C" void kernel_launch(void* const* d_in, const int* in_sizes, int n_in,
                              void* d_out, int out_size, void* d_ws, size_t ws_size,
                              hipStream_t stream) {
  (void)in_sizes; (void)n_in; (void)out_size; (void)ws_size;
  const float* target = (const float*)d_in[0];   // [8,1024,512]
  const float* source = (const float*)d_in[1];   // [1000,4096]
  const float* value  = (const float*)d_in[2];   // [1000,4096]
  const float* Wq = (const float*)d_in[3];
  const float* bq = (const float*)d_in[4];
  const float* Wk = (const float*)d_in[5];
  const float* bk = (const float*)d_in[6];
  const float* Wv = (const float*)d_in[7];
  const float* bv = (const float*)d_in[8];
  const float* Wo = (const float*)d_in[9];
  const float* bo = (const float*)d_in[10];
  float* out = (float*)d_out;

  char* ws = (char*)d_ws;
  __bf16* qws = (__bf16*)ws;                 ws += (size_t)8192 * 1024 * 2;
  __bf16* kws = (__bf16*)ws;                 ws += (size_t)1024 * 1024 * 2;
  __bf16* vws = (__bf16*)ws;                 ws += (size_t)1024 * 1024 * 2;
  __bf16* aws = (__bf16*)ws;                 // attn out [8192,1024]

  const dim3 blk(256);
  const float qscale = 0.08838834764831845f;  // 1/sqrt(128)

  // K/V projections: [1000(pad 1024),4096] @ [4096,1024] -> bf16 (pad rows = bias)
  gemm_bias_kernel<float, true><<<dim3(8, 8), blk, 0, stream>>>(
      source, Wk, bk, kws, 1000, 4096, 1024, 1.0f);
  gemm_bias_kernel<float, true><<<dim3(8, 8), blk, 0, stream>>>(
      value, Wv, bv, vws, 1000, 4096, 1024, 1.0f);
  // Q projection (scale folded in): [8192,512] @ [512,1024]
  gemm_bias_kernel<float, true><<<dim3(8, 64), blk, 0, stream>>>(
      target, Wq, bq, qws, 8192, 512, 1024, qscale);
  // attention: 8 b * 8 h * 8 L-tiles
  attention_kernel<<<dim3(512), blk, 0, stream>>>(qws, kws, vws, aws);
  // output projection: [8192,1024] @ [1024,4096] -> f32 out
  gemm_bias_kernel<__bf16, false><<<dim3(32, 64), blk, 0, stream>>>(
      aws, Wo, bo, out, 8192, 1024, 4096, 1.0f);
}
